// AttentionRNNCell_88373247082750
// MI455X (gfx1250) — matrix operand order repaired and checked by reading, not compiled, not run on hardware
//
#include <hip/hip_runtime.h>
#include <hip/hip_bf16.h>
#include <math.h>

typedef _Float16 f16;
typedef __attribute__((ext_vector_type(16))) _Float16 v16h;
typedef __attribute__((ext_vector_type(8)))  _Float16 v8h;
typedef __attribute__((ext_vector_type(8)))  float    v8f;

#define B_  4
#define N_  2048
#define D_  512
#define H_  8
#define O_  64
#define HO_ 512          // H*O
#define BN  (B_*N_)      // 8192
#define NC  (HO_*3)      // 1536 fused-QKV gemm cols

// ---------------- WMMA + fragment helpers (CDNA5 wave32, 16x16x32 f16 -> f32) ----------------

__device__ __forceinline__ v8f wmma32(v16h a, v16h b, v8f c) {
  return __builtin_amdgcn_wmma_f32_16x16x32_f16(false, a, false, b, (short)0, c, false, false);
}

__device__ __forceinline__ v16h make16(v8h lo, v8h hi) {
  return __builtin_shufflevector(lo, hi, 0,1,2,3,4,5,6,7,8,9,10,11,12,13,14,15);
}

// A fragment (16x32, row-major source, element (m,k) at base[m*ld+k]).
// Per ISA 7.12.2 a lane's 16 halves are k = grp*8+[0,8) and 16+grp*8+[0,8):
// two contiguous 16B runs -> two b128 loads. ld must be a multiple of 8 halves.
__device__ __forceinline__ v16h load_afrag(const f16* base, int ld, int lane) {
  int m = lane & 15, g8 = (lane >> 4) << 3;
  const f16* p = base + (size_t)m * ld + g8;
  return make16(*(const v8h*)p, *(const v8h*)(p + 16));
}
// B fragment (32x16, n-major source, element (k,n) at base[n*ld+k]).
// Lane's 16 halves are k = grp*8+[0,8) and 16+grp*8+[0,8): two contiguous 16B runs.
__device__ __forceinline__ v16h load_bfrag(const f16* base, int ld, int lane) {
  int n = lane & 15, k0 = (lane >> 4) << 3;
  const f16* p = base + (size_t)n * ld + k0;
  return make16(*(const v8h*)p, *(const v8h*)(p + 16));
}

__device__ __forceinline__ float sinpos(int n, int d) {
  // (1/1000)^(2*(d//2)/D); ln(1e-3) = -6.907755279
  float ts  = __expf(-6.9077552790f * ((float)(d & ~1) * (1.0f / (float)D_)));
  float ang = (float)n * ts;
  return (d & 1) ? __cosf(ang) : __sinf(ang);
}
__device__ __forceinline__ float elu1(float x) { return x > 0.f ? x + 1.f : __expf(x); }

// ---------------- Kernel 0a: x = inputs + sine positional encoding (f16) ----------------
__global__ __launch_bounds__(256)
void k_embed(const float* __restrict__ inp, f16* __restrict__ xh) {
  size_t i = (size_t)blockIdx.x * 256 + threadIdx.x;       // < BN*D
  int k = (int)(i & (D_ - 1));
  int gm = (int)(i >> 9);
  int n = gm & (N_ - 1);
  xh[i] = (f16)(inp[i] + sinpos(n, k));
}

// ---------------- Kernel 0b: pre-transpose all B operands to f16 n-major ----------------
// wkT [c][k] (1536x512), memhT [bh][o][d] (32x64x64), woT [o][c] (64x512)
__global__ __launch_bounds__(256)
void k_prep(const float* __restrict__ wk, const float* __restrict__ mem,
            const float* __restrict__ wo,
            f16* __restrict__ wkT, f16* __restrict__ memhT, f16* __restrict__ woT) {
  int i = blockIdx.x * 256 + threadIdx.x;
  if (i < NC * D_) {
    int c = i >> 9, k = i & 511;
    wkT[i] = (f16)wk[(size_t)k * NC + c];
    return;
  }
  int j = i - NC * D_;
  if (j < 32 * O_ * O_) {
    int bh = j >> 12, o = (j >> 6) & 63, d = j & 63;
    memhT[j] = (f16)mem[(size_t)bh * (O_ * O_) + d * O_ + o];
    return;
  }
  int l = j - 32 * O_ * O_;
  if (l < O_ * HO_) {
    int o = l >> 9, c = l & 511, hh = c >> 6, ii = c & 63;
    woT[l] = (f16)wo[(size_t)(ii * H_ + hh) * O_ + o];
  }
}

// ---------------- Kernel 1: fused QKV projection GEMM (no LDS, all b128 fragments) ----------------
// C[8192 x 1536] = xh[8192 x 512] * wkT^T ; scatter into q/k/v (+elu, +transposed copies)
__global__ __launch_bounds__(128)
void k_qkv(const f16* __restrict__ xh, const f16* __restrict__ wkT,
           f16* __restrict__ qh, f16* __restrict__ qeh, f16* __restrict__ kh,
           f16* __restrict__ keth, f16* __restrict__ vth, float* __restrict__ vf) {
  const int tid = threadIdx.x, lane = tid & 31, w = tid >> 5;
  const int m0 = blockIdx.x * 64 + w * 16;
  const int c0 = blockIdx.y * 64;
  v8f acc[4] = {};
  const f16* abase = xh + (size_t)m0 * D_;
  for (int k0 = 0; k0 < D_; k0 += 32) {
    v16h a = load_afrag(abase + k0, D_, lane);
    if (k0 + 32 < D_) __builtin_prefetch(abase + k0 + 32, 0, 1);
#pragma unroll
    for (int ct = 0; ct < 4; ++ct)
      acc[ct] = wmma32(a, load_bfrag(wkT + (size_t)(c0 + ct * 16) * D_ + k0, D_, lane), acc[ct]);
  }
#pragma unroll
  for (int ct = 0; ct < 4; ++ct) {
#pragma unroll
    for (int r = 0; r < 8; ++r) {
      int gm = m0 + r + ((lane >> 4) << 3);
      int gc = c0 + ct * 16 + (lane & 15);
      int s = gc % 3, t = gc / 3;                 // t = h*64+o
      int b = gm >> 11, n = gm & (N_ - 1);
      size_t idx = (size_t)gm * HO_ + t;
      size_t tr  = (size_t)(b * H_ * O_ + t) * N_ + n;   // [bh*64+o][n]
      float v = acc[ct][r];
      if (s == 0)      { kh[idx] = (f16)v; keth[tr] = (f16)elu1(v); }
      else if (s == 1) { vth[tr] = (f16)v; vf[idx] = v; }
      else             { qh[idx] = (f16)v; qeh[idx] = (f16)elu1(v); }
    }
  }
}

// ---------------- Kernel 2: linear-attention readout A_mem, and (v - A_mem) ----------------
__global__ __launch_bounds__(128)
void k_amem(const f16* __restrict__ qeh, const f16* __restrict__ memhT,
            const float* __restrict__ zv, const float* __restrict__ vf,
            float* __restrict__ amem, f16* __restrict__ vmath) {
  __shared__ float dens[4][16];
  const int tid = threadIdx.x, lane = tid & 31, w = tid >> 5;
  const int bh = blockIdx.y, b = bh >> 3, h = bh & 7;
  const int n0 = blockIdx.x * 64 + w * 16;
  const int gm0 = b * N_ + n0;
  const float* zb = zv + (size_t)bh * O_;

  v8f acc[4] = {};
#pragma unroll
  for (int kk = 0; kk < 64; kk += 32) {
    v16h a = load_afrag(qeh + (size_t)gm0 * HO_ + h * O_ + kk, HO_, lane);
#pragma unroll
    for (int ct = 0; ct < 4; ++ct)
      acc[ct] = wmma32(a, load_bfrag(memhT + ((size_t)bh * O_ + ct * 16) * O_ + kk, O_, lane), acc[ct]);
  }
  {   // denominator: qe . z per row, split across the two 16-lane halves
    int row = lane & 15, d0 = (lane >> 4) * 32;
    const f16* qr = qeh + (size_t)(gm0 + row) * HO_ + h * O_ + d0;
    float s = 0.f;
#pragma unroll
    for (int dd = 0; dd < 32; ++dd) s += (float)qr[dd] * zb[d0 + dd];
    s += __shfl_xor(s, 16, 32);
    if (lane < 16) dens[w][row] = s + 1e-8f;
  }
  __syncthreads();
#pragma unroll
  for (int ct = 0; ct < 4; ++ct)
#pragma unroll
    for (int r = 0; r < 8; ++r) {
      int row = r + ((lane >> 4) << 3);
      int gm = gm0 + row, col = ct * 16 + (lane & 15);
      size_t idx = (size_t)gm * HO_ + h * O_ + col;
      float am = acc[ct][r] / dens[w][row];
      amem[idx] = am;
      vmath[((size_t)bh * O_ + col) * N_ + n0 + row] = (f16)(vf[idx] - am);
    }
}

// ---------------- Kernel 3: causal flash-attention + gated mix (LDS only for P transpose) ----------------
__global__ __launch_bounds__(128)
void k_attn(const f16* __restrict__ qh, const f16* __restrict__ kh,
            const f16* __restrict__ vth, const float* __restrict__ amem,
            const float* __restrict__ beta, f16* __restrict__ ah) {
  __shared__ __align__(16) f16 Ps[4][16][40];
  const int tid = threadIdx.x, lane = tid & 31, w = tid >> 5;
  const int bh = blockIdx.y, b = bh >> 3, h = bh & 7;
  const int q0 = blockIdx.x * 64;
  const float rscale = 0.044194173824159216f;   // 1/sqrt(D)

  const f16* qbase = qh + (size_t)(b * N_ + q0 + w * 16) * HO_ + h * O_;
  v16h aQ0 = load_afrag(qbase, HO_, lane);
  v16h aQ1 = load_afrag(qbase + 32, HO_, lane);
  const f16* kbase = kh + (size_t)(b * N_) * HO_ + h * O_;
  const f16* vbase = vth + (size_t)bh * O_ * N_;

  float mst[8], lst[8];
#pragma unroll
  for (int r = 0; r < 8; ++r) { mst[r] = -1e30f; lst[r] = 0.f; }
  v8f acc[4] = {};

  const int nkb = (q0 >> 5) + 2;                // causal: keys <= q0+63
  for (int jb = 0; jb < nkb; ++jb) {
    int k0 = jb * 32;
    // S = Q K^T : kh is key-major = n-major for the B operand, d contiguous
    v8f s0 = {}, s1 = {};
    const f16* kb0 = kbase + (size_t)k0 * HO_;
    s0 = wmma32(aQ0, load_bfrag(kb0,                          HO_, lane), s0);
    s0 = wmma32(aQ1, load_bfrag(kb0 + 32,                     HO_, lane), s0);
    s1 = wmma32(aQ0, load_bfrag(kb0 + (size_t)16 * HO_,       HO_, lane), s1);
    s1 = wmma32(aQ1, load_bfrag(kb0 + (size_t)16 * HO_ + 32,  HO_, lane), s1);

    float p0[8], p1[8], alpha[8];
#pragma unroll
    for (int r = 0; r < 8; ++r) {
      int qrow = q0 + w * 16 + r + ((lane >> 4) << 3);
      int key  = k0 + (lane & 15);
      float x0 = (key      <= qrow) ? s0[r] * rscale : -1e30f;
      float x1 = (key + 16 <= qrow) ? s1[r] * rscale : -1e30f;
      float mx = fmaxf(x0, x1);
#pragma unroll
      for (int msk = 1; msk < 16; msk <<= 1) mx = fmaxf(mx, __shfl_xor(mx, msk, 32));
      float mnew = fmaxf(mst[r], mx);
      alpha[r] = __expf(mst[r] - mnew);
      p0[r] = __expf(x0 - mnew);
      p1[r] = __expf(x1 - mnew);
      float rs = p0[r] + p1[r];
#pragma unroll
      for (int msk = 1; msk < 16; msk <<= 1) rs += __shfl_xor(rs, msk, 32);
      lst[r] = lst[r] * alpha[r] + rs;
      mst[r] = mnew;
    }
#pragma unroll
    for (int ct = 0; ct < 4; ++ct)
#pragma unroll
      for (int r = 0; r < 8; ++r) acc[ct][r] *= alpha[r];
    // transpose P (C layout -> A layout) through per-wave LDS
#pragma unroll
    for (int r = 0; r < 8; ++r) {
      int row = r + ((lane >> 4) << 3), col = lane & 15;
      Ps[w][row][col]      = (f16)p0[r];
      Ps[w][row][col + 16] = (f16)p1[r];
    }
    v16h aP = load_afrag(&Ps[w][0][0], 40, lane);
    // P * V : vth is o-major, key contiguous
#pragma unroll
    for (int ct = 0; ct < 4; ++ct)
      acc[ct] = wmma32(aP, load_bfrag(vbase + (size_t)(ct * 16) * N_ + k0, N_, lane), acc[ct]);
  }

  const float g = 1.f / (1.f + __expf(-beta[0]));
#pragma unroll
  for (int ct = 0; ct < 4; ++ct)
#pragma unroll
    for (int r = 0; r < 8; ++r) {
      int row = r + ((lane >> 4) << 3);
      int gm  = b * N_ + q0 + w * 16 + row;
      int col = ct * 16 + (lane & 15);
      size_t idx = (size_t)gm * HO_ + h * O_ + col;
      float adot = acc[ct][r] / lst[r];
      ah[idx] = (f16)(g * amem[idx] + (1.f - g) * adot);
    }
}

// ---------------- Kernel 4: delta-rule update next_mem = mem + ke^T (v - A_mem), no LDS ----------------
__global__ __launch_bounds__(128)
void k_delta(const f16* __restrict__ keth, const f16* __restrict__ vmath,
             const float* __restrict__ mem, float* __restrict__ nmem) {
  const int tid = threadIdx.x, lane = tid & 31, w = tid >> 5;
  const int bh = blockIdx.x;
  const f16* kb = keth  + ((size_t)bh * O_ + w * 16) * N_;   // [z][n], n contiguous
  const f16* vb = vmath + (size_t)bh * O_ * N_;              // [o][n], n contiguous
  v8f acc[4] = {};
  for (int n0 = 0; n0 < N_; n0 += 32) {
    v16h a = load_afrag(kb + n0, N_, lane);
#pragma unroll
    for (int ct = 0; ct < 4; ++ct)
      acc[ct] = wmma32(a, load_bfrag(vb + (size_t)(ct * 16) * N_ + n0, N_, lane), acc[ct]);
  }
#pragma unroll
  for (int ct = 0; ct < 4; ++ct)
#pragma unroll
    for (int r = 0; r < 8; ++r) {
      int zz = w * 16 + r + ((lane >> 4) << 3);
      int oo = ct * 16 + (lane & 15);
      size_t idx = ((size_t)bh * O_ + zz) * O_ + oo;
      nmem[idx] = mem[idx] + acc[ct][r];
    }
}

// ---------------- Kernel 5: output projection out = A * W, no LDS ----------------
__global__ __launch_bounds__(128)
void k_outproj(const f16* __restrict__ ah, const f16* __restrict__ woT,
               float* __restrict__ out) {
  const int tid = threadIdx.x, lane = tid & 31, w = tid >> 5;
  const int gm0 = blockIdx.x * 64 + w * 16;
  const f16* abase = ah + (size_t)gm0 * HO_;
  v8f acc[4] = {};
  for (int k0 = 0; k0 < HO_; k0 += 32) {
    v16h a = load_afrag(abase + k0, HO_, lane);
    if (k0 + 32 < HO_) __builtin_prefetch(abase + k0 + 32, 0, 1);
#pragma unroll
    for (int ct = 0; ct < 4; ++ct)
      acc[ct] = wmma32(a, load_bfrag(woT + (size_t)(ct * 16) * HO_ + k0, HO_, lane), acc[ct]);
  }
#pragma unroll
  for (int ct = 0; ct < 4; ++ct)
#pragma unroll
    for (int r = 0; r < 8; ++r) {
      int gm = gm0 + r + ((lane >> 4) << 3);
      int col = ct * 16 + (lane & 15);
      out[(size_t)gm * O_ + col] = acc[ct][r];
    }
}

// ---------------- Kernel 6: next_z = z + sum_n elu(k)+1 (contiguous reduce over keth) ----------------
__global__ __launch_bounds__(256)
void k_nextz(const f16* __restrict__ keth, const float* __restrict__ zv,
             float* __restrict__ nz) {
  __shared__ float red[256];
  const int tid = threadIdx.x, bh = blockIdx.x;
  const int d = tid & 63, sl = tid >> 6;
  const f16* src = keth + ((size_t)bh * O_ + d) * N_ + sl * 512;
  float s = 0.f;
  for (int n = 0; n < 512; ++n) s += (float)src[n];
  red[tid] = s;
  __syncthreads();
  if (sl == 0)
    nz[(size_t)bh * O_ + d] = zv[(size_t)bh * O_ + d]
                            + red[d] + red[64 + d] + red[128 + d] + red[192 + d];
}

// ---------------- launch ----------------
extern "C" void kernel_launch(void* const* d_in, const int* in_sizes, int n_in,
                              void* d_out, int out_size, void* d_ws, size_t ws_size,
                              hipStream_t stream) {
  (void)in_sizes; (void)n_in; (void)out_size; (void)ws_size;
  const float* inputs = (const float*)d_in[0];
  const float* mem    = (const float*)d_in[1];
  const float* zv     = (const float*)d_in[2];
  const float* wk     = (const float*)d_in[3];
  const float* wo     = (const float*)d_in[4];
  const float* beta   = (const float*)d_in[5];

  float* out  = (float*)d_out;                     // (B,N,O)     524288
  float* nmem = out + (size_t)BN * O_;             // (B,H,O,O)   131072
  float* nz   = nmem + (size_t)B_ * H_ * O_ * O_;  // (B,H,O)       2048

  const size_t nE = (size_t)BN * HO_;              // 4,194,304 elems per tensor
  f16* xh    = (f16*)d_ws;
  f16* qh    = xh    + nE;
  f16* qeh   = qh    + nE;
  f16* kh    = qeh   + nE;
  f16* keth  = kh    + nE;      // [bh*64+d][n]
  f16* vth   = keth  + nE;      // [bh*64+o][n]
  f16* vmath = vth   + nE;      // [bh*64+o][n]
  f16* ahh   = vmath + nE;
  f16* wkT   = ahh   + nE;                  // 786432
  f16* memhT = wkT   + (size_t)NC * D_;     // 131072
  f16* woT   = memhT + (size_t)32 * O_ * O_;//  32768
  float* vf   = (float*)(woT + (size_t)O_ * HO_);
  float* amem = vf + nE;

  k_embed  <<<(int)(nE / 256), 256, 0, stream>>>(inputs, xh);
  k_prep   <<<(NC * D_ + 32 * O_ * O_ + O_ * HO_ + 255) / 256, 256, 0, stream>>>(
             wk, mem, wo, wkT, memhT, woT);
  k_qkv    <<<dim3(BN / 64, NC / 64), 128, 0, stream>>>(xh, wkT, qh, qeh, kh, keth, vth, vf);
  k_amem   <<<dim3(N_ / 64, B_ * H_), 128, 0, stream>>>(qeh, memhT, zv, vf, amem, vmath);
  k_attn   <<<dim3(N_ / 64, B_ * H_), 128, 0, stream>>>(qh, kh, vth, amem, beta, ahh);
  k_delta  <<<B_ * H_, 128, 0, stream>>>(keth, vmath, mem, nmem);
  k_outproj<<<BN / 64, 128, 0, stream>>>(ahh, woT, out);
  k_nextz  <<<B_ * H_, 256, 0, stream>>>(keth, zv, nz);
}
